// nca_model_57887569215786
// MI455X (gfx1250) — compile-verified, matrix operand order checked by reading, not run on hardware
//
#include <hip/hip_runtime.h>

// Problem constants (from reference): B=4, C=16, D=64, H=128, P=4*C=64
#define CC    16
#define PP    64              // perception channels
#define HH    128             // hidden channels
#define TX    4
#define TY    4
#define TZ    8
#define TILE_V 128            // TX*TY*TZ voxels per workgroup
#define HX    6               // TX+2 halo
#define HY    6
#define HZ    10
#define NHALO (HX*HY*HZ)      // 360
#define NZP   (HX*HY*TZ)      // 288  z-pass outputs
#define NYP   (HX*TY*TZ)      // 192  y-pass outputs
#define PSTR  72              // p_lds row stride in bf16 elems (64 + pad)
#define NW1F  16              // W1 fragments: 8 n-tiles * 2 k-steps
#define NW2F  4               // W2 fragments: 4 k-steps

typedef __attribute__((ext_vector_type(16))) __bf16        v16bf;
typedef __attribute__((ext_vector_type(8)))  float         v8f;
typedef __attribute__((ext_vector_type(8)))  unsigned int  v8u;

// fp32 -> bf16 (round-to-nearest), result in low 16 bits (upper zero)
__device__ __forceinline__ unsigned int f2bf(float f) {
    unsigned int u = __builtin_bit_cast(unsigned int, f);
    return (u + 0x8000u) >> 16;
}
__device__ __forceinline__ unsigned int pack2bf(float lo, float hi) {
    return f2bf(lo) | (f2bf(hi) << 16);
}

// Map fragment slot (j, lane) -> K offset within a 32-wide K tile.
// ISA 16-bit 16x32 layout: lanes 0-15 row=lane, Kbase=0; lanes 16-31
// row=lane-16, Kbase=8; VGPR j<4 -> K=Kbase+2j, j>=4 -> K=Kbase+16+2(j-4).
__device__ __forceinline__ int frag_koff(int j, int lane) {
    return ((lane >> 4) << 3) + ((j < 4) ? 2 * j : 16 + 2 * (j - 4));
}

// Load a 16x32 bf16 fragment from LDS rows (row-major [row][K]) — used for p^T.
__device__ __forceinline__ v16bf load_frag_lds(const unsigned short* base,
                                               int row0, int rstride, int k0) {
    const int lane = threadIdx.x & 31;
    const int m    = lane & 15;
    const int kb   = (lane >> 4) << 3;
    const unsigned short* r = base + (row0 + m) * rstride + k0 + kb;
    v8u u;
#pragma unroll
    for (int j = 0; j < 4; ++j) {
        u[j]     = *(const unsigned int*)(r + 2 * j);
        u[j + 4] = *(const unsigned int*)(r + 16 + 2 * j);
    }
    return __builtin_bit_cast(v16bf, u);
}

// Load a pre-packed fragment from LDS: layout [frag][j][lane] of uints,
// so each ds_load_b32 is bank-conflict-free across the wave.
__device__ __forceinline__ v16bf load_frag_packed(const unsigned int* base, int frag) {
    const int lane = threadIdx.x & 31;
    v8u u;
#pragma unroll
    for (int j = 0; j < 8; ++j)
        u[j] = base[(frag * 8 + j) * 32 + lane];
    return __builtin_bit_cast(v16bf, u);
}

__global__ __launch_bounds__(256)
void nca_step_kernel(const float* __restrict__ x,
                     const float* __restrict__ rand_u,
                     const float* __restrict__ w1,
                     const float* __restrict__ b1,
                     const float* __restrict__ w2,
                     float* __restrict__ out) {
    __shared__ __align__(16) unsigned short p_lds[TILE_V * PSTR];   // 18.0 KB
    __shared__ __align__(16) unsigned int   w1f[NW1F * 8 * 32];     // 16.0 KB
    __shared__ __align__(16) unsigned int   w2f[NW2F * 8 * 32];     //  4.0 KB
    __shared__ __align__(16) float xt[NHALO];                       // halo tile
    __shared__ __align__(16) float pz_g[NZP], pz_d[NZP];            // z-pass
    __shared__ __align__(16) float py_a[NYP], py_b[NYP], py_c[NYP]; // y-pass
    __shared__ float alive_lds[TILE_V];
    __shared__ float smask_lds[TILE_V];

    const int tid = threadIdx.x;

    // ---- tile decomposition: 16 x-tiles * 16 y-tiles * 8 z-tiles per batch
    const int bid = blockIdx.x;
    const int b   = bid >> 11;
    const int rem = bid & 2047;
    const int z0  = (rem & 7) * TZ;
    const int y0  = ((rem >> 3) & 15) * TY;
    const int x0  = ((rem >> 7) & 15) * TX;

    // ---- stage W1/W2 as bf16 fragments in LDS (once per workgroup) -------
    for (int i = tid; i < NW1F * 8 * 32; i += 256) {
        const int lane = i & 31;
        const int j    = (i >> 5) & 7;
        const int frag = i >> 8;                 // nt*2 + ks
        const int row  = (frag >> 1) * 16 + (lane & 15);
        const int K    = (frag & 1) * 32 + frag_koff(j, lane);
        w1f[i] = pack2bf(w1[row * PP + K], w1[row * PP + K + 1]);
    }
    for (int i = tid; i < NW2F * 8 * 32; i += 256) {
        const int lane = i & 31;
        const int j    = (i >> 5) & 7;
        const int kt   = i >> 8;
        const int row  = lane & 15;
        const int K    = kt * 32 + frag_koff(j, lane);
        w2f[i] = pack2bf(w2[row * HH + K], w2[row * HH + K + 1]);
    }

    // ---- stochastic mask
    if (tid < TILE_V) {
        const int v  = tid;
        const int kz = v & 7, jy = (v >> 3) & 3, ix = v >> 5;
        const long spat = ((long)(x0 + ix) << 12) | ((y0 + jy) << 6) | (z0 + kz);
        smask_lds[v] = (rand_u[((long)b << 18) + spat] <= 0.5f) ? 1.f : 0.f;
    }

    // ================= perception: separable sobel via LDS passes ==========
    for (int c = 0; c < CC; ++c) {
        const float* xc = x + ((long)(b * CC + c) << 18);

        // halo load (6x6x10, zero-fill OOB == zero padding for the conv)
        for (int idx = tid; idx < NHALO; idx += 256) {
            const int hz = idx % HZ;
            const int hy = (idx / HZ) % HY;
            const int hx = idx / (HZ * HY);
            const int gx = x0 - 1 + hx, gy = y0 - 1 + hy, gz = z0 - 1 + hz;
            float val = 0.f;
            if ((unsigned)gx < 64u && (unsigned)gy < 64u && (unsigned)gz < 64u)
                val = xc[((long)gx << 12) + (gy << 6) + gz];
            xt[idx] = val;
        }
        __syncthreads();

        // alive mask from raw alpha (channel 3), -inf padding semantics
        if (c == 3 && tid < TILE_V) {
            const int v  = tid;
            const int kz = v & 7, jy = (v >> 3) & 3, ix = v >> 5;
            float amax = -3.4e38f;
#pragma unroll
            for (int di = 0; di < 3; ++di)
#pragma unroll
                for (int dj = 0; dj < 3; ++dj)
#pragma unroll
                    for (int dk = 0; dk < 3; ++dk) {
                        const int gx = x0 - 1 + ix + di;
                        const int gy = y0 - 1 + jy + dj;
                        const int gz = z0 - 1 + kz + dk;
                        if ((unsigned)gx < 64u && (unsigned)gy < 64u && (unsigned)gz < 64u)
                            amax = fmaxf(amax, xt[((ix + di) * HY + jy + dj) * HZ + kz + dk]);
                    }
            alive_lds[v] = (amax > 0.1f) ? 1.f : 0.f;
        }

        // z-pass: g and d along z over the 6x6 xy-halo
        for (int idx = tid; idx < NZP; idx += 256) {
            const int kz = idx & 7;
            const int hy = (idx >> 3) % HY;
            const int hx = idx / (8 * HY);
            const float a0 = xt[(hx * HY + hy) * HZ + kz];
            const float a1 = xt[(hx * HY + hy) * HZ + kz + 1];
            const float a2 = xt[(hx * HY + hy) * HZ + kz + 2];
            pz_g[idx] = a0 + 2.f * a1 + a2;
            pz_d[idx] = a2 - a0;
        }
        __syncthreads();

        // y-pass: (g_y g_z), (d_y g_z), (g_y d_z) over the 6-wide x-halo
        for (int idx = tid; idx < NYP; idx += 256) {
            const int kz = idx & 7;
            const int jy = (idx >> 3) & 3;
            const int hx = idx / (8 * TY);
            const int r0 = (hx * HY + jy) * 8 + kz;
            const float g0 = pz_g[r0], g1 = pz_g[r0 + 8], g2 = pz_g[r0 + 16];
            const float d0 = pz_d[r0], d1 = pz_d[r0 + 8], d2 = pz_d[r0 + 16];
            py_a[idx] = g0 + 2.f * g1 + g2;
            py_b[idx] = g2 - g0;
            py_c[idx] = d0 + 2.f * d1 + d2;
        }
        __syncthreads();

        // x-pass + pack to p_lds (bf16): [ident, sx, sy, sz] at pch = 4c+f
        for (int v = tid; v < TILE_V; v += 256) {
            const int kz = v & 7, jy = (v >> 3) & 3, ix = v >> 5;
            const int r0 = (ix * TY + jy) * 8 + kz;
            const float sx = (py_a[r0 + 64] - py_a[r0]) * 0.125f;
            const float sy = (py_b[r0] + 2.f * py_b[r0 + 32] + py_b[r0 + 64]) * 0.125f;
            const float sz = (py_c[r0] + 2.f * py_c[r0 + 32] + py_c[r0 + 64]) * 0.125f;
            const float ident = xt[((ix + 1) * HY + jy + 1) * HZ + kz + 1];
            unsigned int* pr = (unsigned int*)&p_lds[v * PSTR + (c << 2)];
            pr[0] = pack2bf(ident, sx);
            pr[1] = pack2bf(sy, sz);
        }
        __syncthreads();
    }

    // ============ per-voxel GEMMs via WMMA (wave = 16-voxel N tile) ========
    // GEMM1: h^T[128x16] = relu(W1[128x64] * p^T[64x16] + b1)
    // GEMM2: upd^T[16x16] = W2[16x128] * h^T[128x16]   (h^T never leaves VGPRs)
    const int wave = tid >> 5;
    const int m0   = wave << 4;            // voxel base for this wave
    const int lane = tid & 31;
    const int Ncol = lane & 15;            // voxel-within-tile held by this lane
    const int Mhi  = (lane >> 4) << 3;     // lanes 16-31 hold M rows 8..15

    const v16bf pb0 = load_frag_lds(p_lds, m0, PSTR, 0);    // B: p^T K=0..31
    const v16bf pb1 = load_frag_lds(p_lds, m0, PSTR, 32);   // B: p^T K=32..63

    v8f acc[8];
#pragma unroll
    for (int nt = 0; nt < 8; ++nt) {
        const v16bf wa0 = load_frag_packed(w1f, nt * 2);
        const v16bf wa1 = load_frag_packed(w1f, nt * 2 + 1);
        v8f a = {};
        a = __builtin_amdgcn_wmma_f32_16x16x32_bf16(false, wa0, false, pb0,
                                                    (short)0, a, false, false);
        a = __builtin_amdgcn_wmma_f32_16x16x32_bf16(false, wa1, false, pb1,
                                                    (short)0, a, false, false);
        // bias + relu in-register: lane holds hch = nt*16 + Mhi + r
        const float4 q0 = *(const float4*)(b1 + nt * 16 + Mhi);
        const float4 q1 = *(const float4*)(b1 + nt * 16 + Mhi + 4);
        a[0] = fmaxf(a[0] + q0.x, 0.f);  a[1] = fmaxf(a[1] + q0.y, 0.f);
        a[2] = fmaxf(a[2] + q0.z, 0.f);  a[3] = fmaxf(a[3] + q0.w, 0.f);
        a[4] = fmaxf(a[4] + q1.x, 0.f);  a[5] = fmaxf(a[5] + q1.y, 0.f);
        a[6] = fmaxf(a[6] + q1.z, 0.f);  a[7] = fmaxf(a[7] + q1.w, 0.f);
        acc[nt] = a;
    }

    // GEMM2: B fragments come straight from the accumulators (free transpose):
    // lane needs K = kt*32 + kb + {0..7} from acc[2kt], {16..23} from acc[2kt+1].
    v8f acc2 = {};
#pragma unroll
    for (int kt = 0; kt < 4; ++kt) {
        v8u bu;
#pragma unroll
        for (int j = 0; j < 4; ++j) {
            bu[j]     = pack2bf(acc[2 * kt][2 * j],     acc[2 * kt][2 * j + 1]);
            bu[j + 4] = pack2bf(acc[2 * kt + 1][2 * j], acc[2 * kt + 1][2 * j + 1]);
        }
        const v16bf hb = __builtin_bit_cast(v16bf, bu);
        const v16bf wa = load_frag_packed(w2f, kt);
        acc2 = __builtin_amdgcn_wmma_f32_16x16x32_bf16(false, wa, false, hb,
                                                       (short)0, acc2, false, false);
    }

    // -------- epilogue: out = (x + upd*smask) * alive ----------------------
    // upd^T C-layout: lane holds voxel = m0+Ncol, channels ch = Mhi + r.
    const int  v    = m0 + Ncol;
    const int  kz   = v & 7, jy = (v >> 3) & 3, ix = v >> 5;
    const long spat = ((long)(x0 + ix) << 12) | ((y0 + jy) << 6) | (z0 + kz);
    const float sm  = smask_lds[v];
    const float al  = alive_lds[v];
#pragma unroll
    for (int r = 0; r < 8; ++r) {
        const int  ch  = Mhi + r;
        const long idx = ((long)(b * CC + ch) << 18) + spat;
        out[idx] = (x[idx] + acc2[r] * sm) * al;
    }
}

extern "C" void kernel_launch(void* const* d_in, const int* in_sizes, int n_in,
                              void* d_out, int out_size, void* d_ws, size_t ws_size,
                              hipStream_t stream) {
    const float* x      = (const float*)d_in[0];  // (4,16,64,64,64)
    const float* rand_u = (const float*)d_in[1];  // (4,1,64,64,64)
    const float* w1     = (const float*)d_in[2];  // (128,64)
    const float* b1     = (const float*)d_in[3];  // (128,)
    const float* w2     = (const float*)d_in[4];  // (16,128)
    float* out = (float*)d_out;                   // (4,16,64,64,64)

    dim3 grid(4 * 2048);                          // 4 batches * 16*16*8 tiles
    dim3 block(256);                              // 8 waves (wave32)
    nca_step_kernel<<<grid, block, 0, stream>>>(x, rand_u, w1, b1, w2, out);
}
